// RSSM_2396591751613
// MI455X (gfx1250) — matrix operand order repaired and checked by reading, not compiled, not run on hardware
//
#include <hip/hip_runtime.h>
#include <cstdint>

// ---------------------------------------------------------------------------
// RSSM forward for MI455X (gfx1250): bf16 WMMA GEMMs (fp32 accum), fp32
// elementwise recurrence. Weights transposed to [N,K] bf16 once per call so
// every wave streams contiguous 32B weight chunks (all weights ~86MB bf16,
// L2-resident at 192MB). Wave tile = 64x64 (4x4 WMMA tiles): 16 b128 loads
// per 16 v_wmma_f32_16x16x32_bf16 -> 1.0 loads/WMMA issue ratio.
// ---------------------------------------------------------------------------

#define TT 64
#define BB 256
#define EE 1536
#define AA 6
#define HH 2048
#define LL 1024
#define ALX 128
#define P_IN 2176   // HH + ALX
#define Q_IN 3712   // HH + EE + ALX

typedef __attribute__((ext_vector_type(16))) __bf16 bf16x16;
typedef __attribute__((ext_vector_type(8)))  __bf16 bf16x8;
typedef __attribute__((ext_vector_type(8)))  float  f32x8;

__device__ __forceinline__ float sigm(float x) { return 1.0f / (1.0f + __expf(-x)); }

// ---------------------------------------------------------------------------
// Tiled bf16 WMMA GEMM:  C[M,N] (fp32) = A[M,K](bf16, row stride lda) @ W
// where Wt is the weight stored transposed as [N,K] bf16 (row n contiguous
// along K).  Optional bias[n] add and optional bf16 mirror of C.
// Each wave computes a 64x64 C macro-tile via 4x4 tiles of
// v_wmma_f32_16x16x32_bf16.  Requires M%64==0, N%64==0, K%32==0.
// ---------------------------------------------------------------------------
__global__ void __launch_bounds__(256)
gemm_bf16_wmma(const __bf16* __restrict__ A, int lda,
               const __bf16* __restrict__ Wt,
               const float* __restrict__ bias,
               float* __restrict__ C,
               __bf16* __restrict__ Cbf,
               int M, int N, int K)
{
    const int lane = threadIdx.x & 31;
    const int wave = threadIdx.x >> 5;
    const int tiles_n = N >> 6;                 // 64-wide wave tiles
    const int tiles_m = M >> 6;                 // 64-tall wave tiles
    const int wt = blockIdx.x * 8 + wave;
    if (wt >= tiles_m * tiles_n) return;        // wave-uniform exit
    const int tm = wt / tiles_n;
    const int tn = wt % tiles_n;
    const int row0 = tm << 6;
    const int col0 = tn << 6;
    const int r  = lane & 15;                   // row (A) / col (B,C) within tile
    const int hf = lane >> 4;                   // lane half selects K sub-block

    f32x8 acc[4][4] = {};

    // A: lane holds row (row0+i*16+r); per k-step two 16B chunks at
    // k0 + hf*16 and k0 + hf*16 + 8  (contiguous k, matching B).
    const __bf16* a_base = A + (size_t)(row0 + r) * lda + hf * 16;
    // B: lane holds column (col0+j*16+r); per k-step one contiguous 32B
    // chunk of Wt row:  k0 + hf*16 .. +15.
    const __bf16* b_base = Wt + (size_t)(col0 + r) * (size_t)K + hf * 16;

    for (int k0 = 0; k0 < K; k0 += 32) {
        // prefetch weight stream ahead (global_prefetch_b8)
        __builtin_prefetch(b_base + (size_t)k0 + 256, 0, 1);

        bf16x16 bfr[4];
#pragma unroll
        for (int j = 0; j < 4; ++j)
            bfr[j] = *(const bf16x16*)(b_base + (size_t)j * 16 * K + k0);

        bf16x16 afr[4];
#pragma unroll
        for (int i = 0; i < 4; ++i) {
            const __bf16* ap = a_base + (size_t)i * 16 * lda + k0;
            bf16x8 lo = *(const bf16x8*)(ap);
            bf16x8 hi = *(const bf16x8*)(ap + 8);
#pragma unroll
            for (int e = 0; e < 8; ++e) { afr[i][e] = lo[e]; afr[i][e + 8] = hi[e]; }
        }

#pragma unroll
        for (int i = 0; i < 4; ++i)
#pragma unroll
            for (int j = 0; j < 4; ++j)
                acc[i][j] = __builtin_amdgcn_wmma_f32_16x16x32_bf16(
                    false, afr[i], false, bfr[j], (short)0, acc[i][j], false, false);
    }

    // Epilogue: C tile layout -> lane: col = r, row = hf*8 + g (g = VGPR idx)
#pragma unroll
    for (int i = 0; i < 4; ++i) {
#pragma unroll
        for (int j = 0; j < 4; ++j) {
            const int col = col0 + j * 16 + r;
            const float bv = bias ? bias[col] : 0.0f;
#pragma unroll
            for (int g = 0; g < 8; ++g) {
                const int row = row0 + i * 16 + hf * 8 + g;
                const float v = acc[i][j][g] + bv;
                C[(size_t)row * N + col] = v;
                if (Cbf) Cbf[(size_t)row * N + col] = (__bf16)v;
            }
        }
    }
}

// ---------------------------------------------------------------------------
// fp32 [K,N] -> bf16 [N,K] transpose through LDS (coalesced both sides).
// ---------------------------------------------------------------------------
__global__ void transpose_f32_to_bf16(const float* __restrict__ W,
                                      __bf16* __restrict__ Wt, int K, int N)
{
    __shared__ __bf16 tile[32][33];
    const int kb = blockIdx.y * 32, nb = blockIdx.x * 32;
    const int tx = threadIdx.x, ty = threadIdx.y;   // (32, 8)
    for (int yy = ty; yy < 32; yy += 8) {
        const int k = kb + yy, n = nb + tx;
        tile[yy][tx] = (k < K && n < N) ? (__bf16)W[(size_t)k * N + n] : (__bf16)0.0f;
    }
    __syncthreads();
    for (int yy = ty; yy < 32; yy += 8) {
        const int n = nb + yy, k = kb + tx;
        if (n < N && k < K) Wt[(size_t)n * K + k] = tile[tx][yy];
    }
}

// ---------------------------------------------------------------------------
// Phase-0 state init: h = h0 (fp32), bf16(h0) into heads of xp/xq, z0 -> bf16.
// ---------------------------------------------------------------------------
__global__ void init_state(const float* __restrict__ h0, const float* __restrict__ z0,
                           float* __restrict__ h, __bf16* __restrict__ xp,
                           __bf16* __restrict__ xq, __bf16* __restrict__ z0bf)
{
    const int idx = blockIdx.x * blockDim.x + threadIdx.x;
    if (idx < BB * HH) {
        const int b = idx / HH, j = idx % HH;
        const float v = h0[idx];
        h[idx] = v;
        xp[(size_t)b * P_IN + j] = (__bf16)v;
        xq[(size_t)b * Q_IN + j] = (__bf16)v;
    } else if (idx < BB * HH + BB * LL) {
        const int k = idx - BB * HH;
        z0bf[k] = (__bf16)z0[k];
    }
}

// ---------------------------------------------------------------------------
// Per-step concat tails: pa_t -> xp[:,H:], qa_t -> xq[:,H+E:], emb_t -> xq[:,H:H+E]
// (action layers are tiny A=6 dot products, done inline in fp32).
// ---------------------------------------------------------------------------
__global__ void pack_step(const float* __restrict__ act_t,   // [B,A]
                          const float* __restrict__ emb_t,   // [B,E]
                          const float* __restrict__ W_pa, const float* __restrict__ b_pa,
                          const float* __restrict__ W_qa, const float* __restrict__ b_qa,
                          __bf16* __restrict__ xp, __bf16* __restrict__ xq)
{
    const int idx = blockIdx.x * blockDim.x + threadIdx.x;
    const int n_pa = BB * ALX;
    if (idx < n_pa) {
        const int b = idx / ALX, l = idx % ALX;
        float s = b_pa[l];
        const float* ac = act_t + (size_t)b * AA;
#pragma unroll
        for (int a = 0; a < AA; ++a) s += ac[a] * W_pa[a * ALX + l];
        xp[(size_t)b * P_IN + HH + l] = (__bf16)s;
    } else if (idx < 2 * n_pa) {
        const int k = idx - n_pa;
        const int b = k / ALX, l = k % ALX;
        float s = b_qa[l];
        const float* ac = act_t + (size_t)b * AA;
#pragma unroll
        for (int a = 0; a < AA; ++a) s += ac[a] * W_qa[a * ALX + l];
        xq[(size_t)b * Q_IN + HH + EE + l] = (__bf16)s;
    } else if (idx < 2 * n_pa + BB * EE) {
        const int k = idx - 2 * n_pa;
        const int b = k / EE, e = k % EE;
        xq[(size_t)b * Q_IN + HH + e] = (__bf16)emb_t[k];
    }
}

// ---------------------------------------------------------------------------
// Fused GRU elementwise update. gx/gh already contain b_ih/b_hh.
// Writes h (in place), fp32 h into d_out slab, bf16 h into xp/xq heads.
// ---------------------------------------------------------------------------
__global__ void gru_step(const float* __restrict__ gx, const float* __restrict__ gh,
                         float* __restrict__ h, float* __restrict__ h_out,
                         __bf16* __restrict__ xp, __bf16* __restrict__ xq)
{
    const int idx = blockIdx.x * blockDim.x + threadIdx.x;
    if (idx >= BB * HH) return;
    const int b = idx / HH, j = idx % HH;
    const size_t o = (size_t)b * (3 * HH) + j;
    const float r = sigm(gx[o] + gh[o]);
    const float z = sigm(gx[o + HH] + gh[o + HH]);
    const float n = tanhf(gx[o + 2 * HH] + r * gh[o + 2 * HH]);
    const float hn = (1.0f - z) * n + z * h[idx];
    h[idx] = hn;
    h_out[idx] = hn;
    xp[(size_t)b * P_IN + j] = (__bf16)hn;
    xq[(size_t)b * Q_IN + j] = (__bf16)hn;
}

// ---------------------------------------------------------------------------
// Representation-model head: y=[mu|log_sigma], sigma = .9*sigmoid(ls/.9)+.1,
// samp = mu + noise*sigma.
// ---------------------------------------------------------------------------
__global__ void rep_out(const float* __restrict__ y,       // [B, 2L]
                        const float* __restrict__ noise_t, // [B, L]
                        float* __restrict__ mu_o, float* __restrict__ sig_o,
                        float* __restrict__ samp_o)
{
    const int idx = blockIdx.x * blockDim.x + threadIdx.x;
    if (idx >= BB * LL) return;
    const int b = idx / LL, j = idx % LL;
    const float mu = y[(size_t)b * (2 * LL) + j];
    const float ls = y[(size_t)b * (2 * LL) + LL + j];
    const float sg = 0.9f * sigm(ls * (1.0f / 0.9f)) + 0.1f;
    mu_o[idx] = mu;
    sig_o[idx] = sg;
    samp_o[idx] = mu + noise_t[idx] * sg;
}

// ---------------------------------------------------------------------------
extern "C" void kernel_launch(void* const* d_in, const int* in_sizes, int n_in,
                              void* d_out, int out_size, void* d_ws, size_t ws_size,
                              hipStream_t stream)
{
    (void)in_sizes; (void)n_in; (void)out_size; (void)ws_size;
    const float* embeds      = (const float*)d_in[0];
    const float* actions     = (const float*)d_in[1];
    const float* h0          = (const float*)d_in[2];
    const float* z0          = (const float*)d_in[3];
    const float* noise_prior = (const float*)d_in[4];
    const float* noise_post  = (const float*)d_in[5];
    const float* W_pre = (const float*)d_in[6];
    const float* b_pre = (const float*)d_in[7];
    const float* W_ih  = (const float*)d_in[8];
    const float* W_hh  = (const float*)d_in[9];
    const float* b_ih  = (const float*)d_in[10];
    const float* b_hh  = (const float*)d_in[11];
    const float* W_pa  = (const float*)d_in[12];
    const float* b_pa  = (const float*)d_in[13];
    const float* W_qa  = (const float*)d_in[14];
    const float* b_qa  = (const float*)d_in[15];
    const float* pW1 = (const float*)d_in[16];
    const float* pb1 = (const float*)d_in[17];
    const float* pW2 = (const float*)d_in[18];
    const float* pb2 = (const float*)d_in[19];
    const float* qW1 = (const float*)d_in[20];
    const float* qb1 = (const float*)d_in[21];
    const float* qW2 = (const float*)d_in[22];
    const float* qb2 = (const float*)d_in[23];

    // ---- workspace carve-out (256B aligned) ------------------------------
    char* wsb = (char*)d_ws;
    size_t off = 0;
    auto carve = [&](size_t bytes) -> char* {
        char* p = wsb + off;
        off += (bytes + 255) & ~(size_t)255;
        return p;
    };
    __bf16* Wt_pre = (__bf16*)carve((size_t)LL * HH * 2);
    __bf16* Wt_ih  = (__bf16*)carve((size_t)HH * 3 * HH * 2);
    __bf16* Wt_hh  = (__bf16*)carve((size_t)HH * 3 * HH * 2);
    __bf16* pW1t   = (__bf16*)carve((size_t)P_IN * P_IN * 2);
    __bf16* pW2t   = (__bf16*)carve((size_t)P_IN * 2 * LL * 2);
    __bf16* qW1t   = (__bf16*)carve((size_t)Q_IN * Q_IN * 2);
    __bf16* qW2t   = (__bf16*)carve((size_t)Q_IN * 2 * LL * 2);
    __bf16* z0bf   = (__bf16*)carve((size_t)BB * LL * 2);
    float*  in32   = (float*) carve((size_t)BB * HH * 4);
    __bf16* intbf  = (__bf16*)carve((size_t)BB * HH * 2);
    float*  gx     = (float*) carve((size_t)BB * 3 * HH * 4);
    float*  gh     = (float*) carve((size_t)BB * 3 * HH * 4);
    float*  h      = (float*) carve((size_t)BB * HH * 4);
    __bf16* xp     = (__bf16*)carve((size_t)BB * P_IN * 2);
    __bf16* xq     = (__bf16*)carve((size_t)BB * Q_IN * 2);
    float*  t1p    = (float*) carve((size_t)BB * P_IN * 4);
    __bf16* t1pbf  = (__bf16*)carve((size_t)BB * P_IN * 2);
    float*  t1q    = (float*) carve((size_t)BB * Q_IN * 4);
    __bf16* t1qbf  = (__bf16*)carve((size_t)BB * Q_IN * 2);
    float*  yp     = (float*) carve((size_t)BB * 2 * LL * 4);
    float*  yq     = (float*) carve((size_t)BB * 2 * LL * 4);

    // ---- output slabs (return order: h, p_mu, p_sig, p_samp, q_mu, q_sig, q_samp)
    float* out = (float*)d_out;
    const size_t slabH = (size_t)TT * BB * HH;
    const size_t slabL = (size_t)TT * BB * LL;
    float* out_h     = out;
    float* out_pmu   = out_h + slabH;
    float* out_psig  = out_pmu + slabL;
    float* out_psamp = out_psig + slabL;
    float* out_qmu   = out_psamp + slabL;
    float* out_qsig  = out_qmu + slabL;
    float* out_qsamp = out_qsig + slabL;

    // ---- helpers ---------------------------------------------------------
    auto launch_tr = [&](const float* W, __bf16* Wt, int K, int N) {
        dim3 b(32, 8), g((N + 31) / 32, (K + 31) / 32);
        transpose_f32_to_bf16<<<g, b, 0, stream>>>(W, Wt, K, N);
    };
    auto gemm = [&](const __bf16* Ain, int lda, const __bf16* Wt, const float* bias,
                    float* C, __bf16* Cbf, int M, int N, int K) {
        const int tiles = (M / 64) * (N / 64);
        gemm_bf16_wmma<<<(tiles + 7) / 8, 256, 0, stream>>>(Ain, lda, Wt, bias, C, Cbf, M, N, K);
    };

    // ---- phase 0: weight prep + hoisted GEMMs ----------------------------
    launch_tr(W_pre, Wt_pre, LL, HH);
    launch_tr(W_ih,  Wt_ih,  HH, 3 * HH);
    launch_tr(W_hh,  Wt_hh,  HH, 3 * HH);
    launch_tr(pW1,   pW1t,   P_IN, P_IN);
    launch_tr(pW2,   pW2t,   P_IN, 2 * LL);
    launch_tr(qW1,   qW1t,   Q_IN, Q_IN);
    launch_tr(qW2,   qW2t,   Q_IN, 2 * LL);

    init_state<<<(BB * HH + BB * LL + 255) / 256, 256, 0, stream>>>(h0, z0, h, xp, xq, z0bf);

    // in_t = z0 @ W_pre + b_pre ; gx = in_t @ W_ih + b_ih  (time-invariant)
    gemm(z0bf, LL, Wt_pre, b_pre, in32, intbf, BB, HH, LL);
    gemm(intbf, HH, Wt_ih, b_ih, gx, nullptr, BB, 3 * HH, HH);

    // ---- recurrence ------------------------------------------------------
    const int pack_n = 2 * BB * ALX + BB * EE;
    for (int t = 0; t < TT; ++t) {
        pack_step<<<(pack_n + 255) / 256, 256, 0, stream>>>(
            actions + (size_t)t * BB * AA, embeds + (size_t)t * BB * EE,
            W_pa, b_pa, W_qa, b_qa, xp, xq);

        // gh = h_{t-1} @ W_hh + b_hh  (A = first H cols of xp, lda = P_IN)
        gemm(xp, P_IN, Wt_hh, b_hh, gh, nullptr, BB, 3 * HH, HH);

        gru_step<<<(BB * HH + 255) / 256, 256, 0, stream>>>(
            gx, gh, h, out_h + (size_t)t * BB * HH, xp, xq);

        // prior: y = (xp @ pW1 + pb1) @ pW2 + pb2   (LeakyReLU(1.0) = identity)
        gemm(xp, P_IN, pW1t, pb1, t1p, t1pbf, BB, P_IN, P_IN);
        gemm(t1pbf, P_IN, pW2t, pb2, yp, nullptr, BB, 2 * LL, P_IN);
        rep_out<<<(BB * LL + 255) / 256, 256, 0, stream>>>(
            yp, noise_prior + (size_t)t * BB * LL,
            out_pmu + (size_t)t * BB * LL, out_psig + (size_t)t * BB * LL,
            out_psamp + (size_t)t * BB * LL);

        // posterior
        gemm(xq, Q_IN, qW1t, qb1, t1q, t1qbf, BB, Q_IN, Q_IN);
        gemm(t1qbf, Q_IN, qW2t, qb2, yq, nullptr, BB, 2 * LL, Q_IN);
        rep_out<<<(BB * LL + 255) / 256, 256, 0, stream>>>(
            yq, noise_post + (size_t)t * BB * LL,
            out_qmu + (size_t)t * BB * LL, out_qsig + (size_t)t * BB * LL,
            out_qsamp + (size_t)t * BB * LL);
    }
}
